// shap_normalGRUpertime_74938589380825
// MI455X (gfx1250) — compile-verified, matrix operand order checked by reading, not run on hardware
//
#include <hip/hip_runtime.h>
#include <hip/hip_bf16.h>

typedef __attribute__((ext_vector_type(16))) _Float16 v16h;
typedef __attribute__((ext_vector_type(8)))  _Float16 v8h;
typedef __attribute__((ext_vector_type(8)))  float    v8f;
typedef __attribute__((ext_vector_type(4)))  float    v4f;

constexpr int Bn = 512, Tn = 512, Dn = 128, Hn = 256, On = 2;
constexpr int Rrows = Bn * Tn;            // 262144 flattened (b,t) rows

// ---------------------------------------------------------------------------
// Fragment loader: 16-bit A-operand (and pre-transposed B-operand) layout.
// lane&15 -> row (M for A, N-column for B with Wt = W^T), lane>>4 -> +8 K slab.
// halves 0..7 = K .. K+7 ; halves 8..15 = K+16 .. K+23  (16-byte loads).
// ---------------------------------------------------------------------------
__device__ __forceinline__ v16h ld_frag(const _Float16* p) {
  v8h lo = *(const v8h*)p;
  v8h hi = *(const v8h*)(p + 16);
  return __builtin_shufflevector(lo, hi, 0,1,2,3,4,5,6,7,8,9,10,11,12,13,14,15);
}

__device__ __forceinline__ v8f wmma16(v16h a, v16h b, v8f c) {
  return __builtin_amdgcn_wmma_f32_16x16x32_f16(false, a, false, b, (short)0, c,
                                                false, false);
}

__device__ __forceinline__ float sigf(float x) {
  return 1.0f / (1.0f + __expf(-x));
}

// ---------------------------------------------------------------------------
// Kernel 0: convert U (D,H) -> Ut f16 (H rows, D cols) and W (H,H) -> Wt f16
// (transposed) so WMMA B-fragments are contiguous per lane.
// ---------------------------------------------------------------------------
__global__ __launch_bounds__(256) void gru_prep_kernel(
    const float* __restrict__ Ur, const float* __restrict__ Uz,
    const float* __restrict__ Uh, const float* __restrict__ Wr,
    const float* __restrict__ Wz, const float* __restrict__ Wh,
    _Float16* __restrict__ Ut, _Float16* __restrict__ Wt) {
  int idx = blockIdx.x * 256 + threadIdx.x;
  if (idx < 3 * Hn * Dn) {                       // 98304 Ut elements
    int g = idx / (Hn * Dn);
    int e = idx % (Hn * Dn);
    int n = e >> 7, k = e & 127;                 // Ut[n][k] = U[k][n]
    const float* U = (g == 0) ? Ur : (g == 1) ? Uz : Uh;
    Ut[(size_t)g * (Hn * Dn) + n * Dn + k] = (_Float16)U[(size_t)k * Hn + n];
  }
  if (idx < 3 * Hn * Hn) {                       // 196608 Wt elements
    int g = idx / (Hn * Hn);
    int e = idx % (Hn * Hn);
    int n = e >> 8, k = e & 255;                 // Wt[n][k] = W[k][n]
    const float* W = (g == 0) ? Wr : (g == 1) ? Wz : Wh;
    Wt[(size_t)g * (Hn * Hn) + n * Hn + k] = (_Float16)W[(size_t)k * Hn + n];
  }
}

// ---------------------------------------------------------------------------
// Kernel 1: fused input projections  xr/xz/xh = x @ U_g + b_g ([R,128]x[128,256])
// Block: 256 thr (8 waves), M-block = 32 rows. Wave w covers N = [32w, 32w+32).
// Per wave: 2 M-tiles x 2 N-tiles x 3 gates, K-loop of 4 (K=128).
// Outputs stored non-temporally (written once, read once much later).
// ---------------------------------------------------------------------------
constexpr int XS_STRIDE = 136;                   // 128 + 8 halves pad

__global__ __launch_bounds__(256) void gru_proj_kernel(
    const float* __restrict__ x, const _Float16* __restrict__ Ut,
    const float* __restrict__ br, const float* __restrict__ bz,
    const float* __restrict__ bh, float* __restrict__ xr,
    float* __restrict__ xz, float* __restrict__ xh) {
  __shared__ __align__(16) _Float16 xs[32 * XS_STRIDE];

  const int tid = threadIdx.x;
  const int r0  = blockIdx.x * 32;

  // cooperative load: 32x128 f32 tile -> f16 LDS (16 elems / thread, row-contig)
  {
    int row = tid >> 3;
    int col = (tid & 7) * 16;
    const v4f* src = (const v4f*)(x + (size_t)(r0 + row) * Dn + col);
    v4f f0 = src[0], f1 = src[1], f2 = src[2], f3 = src[3];
    v8h h0 = {(_Float16)f0[0], (_Float16)f0[1], (_Float16)f0[2], (_Float16)f0[3],
              (_Float16)f1[0], (_Float16)f1[1], (_Float16)f1[2], (_Float16)f1[3]};
    v8h h1 = {(_Float16)f2[0], (_Float16)f2[1], (_Float16)f2[2], (_Float16)f2[3],
              (_Float16)f3[0], (_Float16)f3[1], (_Float16)f3[2], (_Float16)f3[3]};
    *(v8h*)&xs[row * XS_STRIDE + col]     = h0;
    *(v8h*)&xs[row * XS_STRIDE + col + 8] = h1;
  }
  __syncthreads();

  const int lane = tid & 31;
  const int w    = tid >> 5;
  const int lr   = lane & 15;
  const int lhi  = lane >> 4;
  const int n0   = w * 32;

  v8f acc[3][2][2];
#pragma unroll
  for (int g = 0; g < 3; ++g)
#pragma unroll
    for (int mi = 0; mi < 2; ++mi)
#pragma unroll
      for (int nj = 0; nj < 2; ++nj) acc[g][mi][nj] = v8f{0,0,0,0,0,0,0,0};

#pragma unroll
  for (int k0 = 0; k0 < Dn; k0 += 32) {
    v16h a0 = ld_frag(&xs[(0  + lr) * XS_STRIDE + k0 + lhi * 8]);
    v16h a1 = ld_frag(&xs[(16 + lr) * XS_STRIDE + k0 + lhi * 8]);
#pragma unroll
    for (int g = 0; g < 3; ++g) {
      const _Float16* Ug = Ut + (size_t)g * (Hn * Dn);
#pragma unroll
      for (int nj = 0; nj < 2; ++nj) {
        v16h b = ld_frag(Ug + (size_t)(n0 + nj * 16 + lr) * Dn + k0 + lhi * 8);
        acc[g][0][nj] = wmma16(a0, b, acc[g][0][nj]);
        acc[g][1][nj] = wmma16(a1, b, acc[g][1][nj]);
      }
    }
  }

#pragma unroll
  for (int g = 0; g < 3; ++g) {
    const float* bias = (g == 0) ? br : (g == 1) ? bz : bh;
    float*       out  = (g == 0) ? xr : (g == 1) ? xz : xh;
#pragma unroll
    for (int nj = 0; nj < 2; ++nj) {
      int   n  = n0 + nj * 16 + lr;
      float bv = bias[n];
#pragma unroll
      for (int mi = 0; mi < 2; ++mi)
#pragma unroll
        for (int i = 0; i < 8; ++i) {
          int row = r0 + mi * 16 + lhi * 8 + i;
          __builtin_nontemporal_store(acc[g][mi][nj][i] + bv,
                                      &out[(size_t)row * Hn + n]);
        }
    }
  }
}

// ---------------------------------------------------------------------------
// Kernel 2: recurrent scan. 32 blocks (16 batch rows each), 256 thr = 8 waves.
// Each wave owns N-chunk [32w, 32w+32); its slice of h (f32) lives in
// REGISTERS for all 512 steps. W_r/W_z are pinned in LDS (f16, padded
// stride); W_h streams from L2. Per step: {r,z} GEMMs -> sigmoid -> h*r (f16
// to LDS) ; barrier ; g GEMM -> tanh -> h update (regs + f16 LDS) ; barrier.
// Next-step gate inputs are prefetched (global_prefetch_b8).
// ---------------------------------------------------------------------------
constexpr int HS16 = 264;                        // f16 row stride for h frags
constexpr int WLS  = 264;                        // f16 row stride for LDS W
constexpr int HFS  = 260;                        // f32 row stride for final h

__global__ __launch_bounds__(256) void gru_scan_kernel(
    const float* __restrict__ xr, const float* __restrict__ xz,
    const float* __restrict__ xh, const _Float16* __restrict__ Wt,
    const float* __restrict__ Wcls, const float* __restrict__ bcls,
    float* __restrict__ out) {
  __shared__ __align__(16) _Float16 sWr[Hn * WLS];   // 132 KB
  __shared__ __align__(16) _Float16 sWz[Hn * WLS];   // 132 KB
  __shared__ __align__(16) _Float16 hA[16 * HS16];   // h   (A-frag source)
  __shared__ __align__(16) _Float16 hrA[16 * HS16];  // h*r (A-frag source)
  __shared__ __align__(16) float    hF[16 * HFS];    // final h for classifier

  const int tid = threadIdx.x;
  const _Float16* WtR = Wt;
  const _Float16* WtZ = Wt + (size_t)(Hn * Hn);
  const _Float16* WtH = Wt + (size_t)(2 * Hn * Hn);

  // preload W_r, W_z into LDS (v8h granularity, 32 iters each)
  for (int idx = tid; idx < Hn * (Hn / 8); idx += 256) {
    int row = idx >> 5;
    int col = (idx & 31) * 8;
    *(v8h*)&sWr[row * WLS + col] = *(const v8h*)(WtR + (size_t)row * Hn + col);
    *(v8h*)&sWz[row * WLS + col] = *(const v8h*)(WtZ + (size_t)row * Hn + col);
  }
  for (int i = tid; i < 16 * HS16; i += 256) hA[i] = (_Float16)0.0f;
  __syncthreads();

  const int b0   = blockIdx.x * 16;
  const int lane = tid & 31;
  const int w    = tid >> 5;
  const int lr   = lane & 15;
  const int lhi  = lane >> 4;
  const int n0   = w * 32;

  v8f hv[2] = {v8f{0,0,0,0,0,0,0,0}, v8f{0,0,0,0,0,0,0,0}};  // h, in regs

  for (int t = 0; t < Tn; ++t) {
    // ---- phase 1: r,z gates (W_r, W_z from LDS) ----------------------------
    v8f cr[2] = {v8f{0,0,0,0,0,0,0,0}, v8f{0,0,0,0,0,0,0,0}};
    v8f cz[2] = {v8f{0,0,0,0,0,0,0,0}, v8f{0,0,0,0,0,0,0,0}};
#pragma unroll
    for (int k0 = 0; k0 < Hn; k0 += 32) {
      v16h a = ld_frag(&hA[lr * HS16 + k0 + lhi * 8]);
#pragma unroll
      for (int nj = 0; nj < 2; ++nj) {
        const int brow = (n0 + nj * 16 + lr) * WLS + k0 + lhi * 8;
        v16h bR = ld_frag(&sWr[brow]);
        v16h bZ = ld_frag(&sWz[brow]);
        cr[nj] = wmma16(a, bR, cr[nj]);
        cz[nj] = wmma16(a, bZ, cz[nj]);
      }
    }

    v8f zv[2];
#pragma unroll
    for (int nj = 0; nj < 2; ++nj) {
      const int n = n0 + nj * 16 + lr;
#pragma unroll
      for (int i = 0; i < 8; ++i) {
        const int    m  = lhi * 8 + i;
        const size_t ix = ((size_t)(b0 + m) * Tn + t) * Hn + n;
        float rr  = sigf(cr[nj][i] + __builtin_nontemporal_load(&xr[ix]));
        zv[nj][i] = sigf(cz[nj][i] + __builtin_nontemporal_load(&xz[ix]));
        hrA[m * HS16 + n] = (_Float16)(hv[nj][i] * rr);
      }
    }
    __syncthreads();

    // ---- phase 2: candidate + state update (W_h from L2) -------------------
    v8f cg[2] = {v8f{0,0,0,0,0,0,0,0}, v8f{0,0,0,0,0,0,0,0}};
#pragma unroll
    for (int k0 = 0; k0 < Hn; k0 += 32) {
      v16h a = ld_frag(&hrA[lr * HS16 + k0 + lhi * 8]);
#pragma unroll
      for (int nj = 0; nj < 2; ++nj) {
        v16h bH = ld_frag(WtH + (size_t)(n0 + nj * 16 + lr) * Hn + k0 + lhi * 8);
        cg[nj] = wmma16(a, bH, cg[nj]);
      }
    }
#pragma unroll
    for (int nj = 0; nj < 2; ++nj) {
      const int n = n0 + nj * 16 + lr;
#pragma unroll
      for (int i = 0; i < 8; ++i) {
        const int    m  = lhi * 8 + i;
        const size_t ix = ((size_t)(b0 + m) * Tn + t) * Hn + n;
        float g    = tanhf(cg[nj][i] + __builtin_nontemporal_load(&xh[ix]));
        float z    = zv[nj][i];
        float hnew = z * hv[nj][i] + (1.0f - z) * g;
        hv[nj][i]  = hnew;
        hA[m * HS16 + n] = (_Float16)hnew;
        if (nj == 0) {                       // warm t+1 inputs (one line / row)
          __builtin_prefetch(&xr[ix + Hn], 0, 1);
          __builtin_prefetch(&xz[ix + Hn], 0, 1);
          __builtin_prefetch(&xh[ix + Hn], 0, 1);
        }
      }
    }
    __syncthreads();
  }

  // ---- classifier: h_final @ W_cls + b_cls  (O=2, tiny) ---------------------
#pragma unroll
  for (int nj = 0; nj < 2; ++nj) {
    const int n = n0 + nj * 16 + lr;
#pragma unroll
    for (int i = 0; i < 8; ++i) hF[(lhi * 8 + i) * HFS + n] = hv[nj][i];
  }
  __syncthreads();
  if (tid < 32) {
    const int row = tid >> 1;
    const int o   = tid & 1;
    float acc = bcls[o];
    for (int k = 0; k < Hn; ++k) acc += hF[row * HFS + k] * Wcls[k * On + o];
    out[(size_t)(b0 + row) * On + o] = acc;
  }
}

// ---------------------------------------------------------------------------
extern "C" void kernel_launch(void* const* d_in, const int* in_sizes, int n_in,
                              void* d_out, int out_size, void* d_ws, size_t ws_size,
                              hipStream_t stream) {
  const float* x   = (const float*)d_in[0];
  const float* Ur  = (const float*)d_in[1];
  const float* Wr  = (const float*)d_in[2];
  const float* br  = (const float*)d_in[3];
  const float* Uz  = (const float*)d_in[4];
  const float* Wz  = (const float*)d_in[5];
  const float* bz  = (const float*)d_in[6];
  const float* Uh  = (const float*)d_in[7];
  const float* Wh  = (const float*)d_in[8];
  const float* bh  = (const float*)d_in[9];
  const float* Wc  = (const float*)d_in[10];
  const float* bc  = (const float*)d_in[11];
  float* out = (float*)d_out;

  const size_t SX = (size_t)Rrows * Hn;            // elems per projection
  char* ws = (char*)d_ws;
  float* xr = (float*)ws;
  float* xz = xr + SX;
  float* xh = xz + SX;
  _Float16* Ut = (_Float16*)(ws + 3 * SX * sizeof(float));
  _Float16* Wt = Ut + (size_t)3 * Hn * Dn;

  gru_prep_kernel<<<768, 256, 0, stream>>>(Ur, Uz, Uh, Wr, Wz, Wh, Ut, Wt);
  gru_proj_kernel<<<Rrows / 32, 256, 0, stream>>>(x, Ut, br, bz, bh, xr, xz, xh);
  gru_scan_kernel<<<Bn / 16, 256, 0, stream>>>(xr, xz, xh, Wt, Wc, bc, out);
}
